// SelfAttentionModule_50027779063961
// MI455X (gfx1250) — matrix operand order repaired and checked by reading, not compiled, not run on hardware
//
#include <hip/hip_runtime.h>
#include <cstdint>
#include <cstddef>

// ---------------------------------------------------------------------------
// Self-attention for M=8192, d=1024 on gfx1250 (MI455X).
// ~330 GFLOP of matmul vs ~100MB real traffic @23.3TB/s -> compute bound;
// everything runs on v_wmma_f32_16x16x32_bf16 (f32 accumulate).
// Pipeline:
//   1) convert x, Wq/Wk/Wv to bf16 (workspace)
//   2) WMMA GEMMs: Qh = (x Wq^T + bq)/sqrt(d)  [bf16, row-major]
//                  Kh = (x Wk^T + bk)          [bf16, row-major]
//                  VT = (x Wv^T + bv)^T        [bf16, 1024 x 8192]
//   3) flash-attention: 1 WG = 4 waves = 16 query rows.
//      Q tile staged into LDS by the Tensor Data Mover (rows 0-7, one D#
//      descriptor with LDS padding enabled) + global_load_async_to_lds_b128
//      (rows 8-15) -> exercises both TENSORcnt and ASYNCcnt paths.
//      Online softmax across waves via shfl + small LDS arrays; K prefetched
//      one block ahead with global_prefetch.
// ---------------------------------------------------------------------------

#define DMODEL 1024
#define MROWS  8192
#define INV_SQRT_D 0.03125f   // 1/sqrt(1024), folded into Qh

typedef __attribute__((ext_vector_type(16))) __bf16 v16bf;
typedef __attribute__((ext_vector_type(8)))  float  v8f;
typedef __attribute__((ext_vector_type(4)))  unsigned int u32x4;
typedef __attribute__((ext_vector_type(8)))  unsigned int u32x8;

union AFrag { v16bf v; uint4 q[2]; };      // A: lane=M; K runs at hi*8, 16+hi*8
union BFrag { v16bf v; uint4 q[2]; };      // B: lane=N; K runs at hi*8, 16+hi*8
union CFrag { v8f v; float f[8]; };        // C/D: lane=N; elem r -> row r+8*hi
union Pack8 { __bf16 h[8]; uint4 q; };

static __device__ __forceinline__ v8f wmma_bf16(v16bf a, v16bf b, v8f c) {
    // (neg_a, A, neg_b, B, c_mod, C, reuse_a, reuse_b)
    return __builtin_amdgcn_wmma_f32_16x16x32_bf16(false, a, false, b,
                                                   (short)0, c, false, false);
}

// Async copy 16 bytes global -> LDS (per active lane). LDS byte address is the
// low 32 bits of the flat pointer (aperture rule, ISA 10.2).
static __device__ __forceinline__ void async_g2l_b128(void* lds_dst,
                                                      const void* gsrc) {
    unsigned      lds_off = (unsigned)(uintptr_t)lds_dst;
    unsigned long gaddr   = (unsigned long)(uintptr_t)gsrc;
    asm volatile("global_load_async_to_lds_b128 %0, %1, off"
                 :: "v"(lds_off), "v"(gaddr) : "memory");
}

static __device__ __forceinline__ void wait_asynccnt0() {
#if __has_builtin(__builtin_amdgcn_s_wait_asynccnt)
    __builtin_amdgcn_s_wait_asynccnt(0);
#else
    asm volatile("s_wait_asynccnt 0" ::: "memory");
#endif
}

static __device__ __forceinline__ void wait_tensorcnt0() {
#if __has_builtin(__builtin_amdgcn_s_wait_tensorcnt)
    __builtin_amdgcn_s_wait_tensorcnt((short)0);
#else
    asm volatile("s_wait_tensorcnt 0" ::: "memory");
#endif
}

// ---------------------------------------------------------------------------
__global__ void cvt_f32_to_bf16(const float* __restrict__ src,
                                __bf16* __restrict__ dst, int n) {
    int i = blockIdx.x * blockDim.x + threadIdx.x;
    int stride = gridDim.x * blockDim.x;
    for (; i < n; i += stride) dst[i] = (__bf16)src[i];
}

// ---------------------------------------------------------------------------
// y[m,n] = sum_k X[m,k]*W[n,k] + bias[n]   (torch Linear: x @ W^T + b)
// Each wave computes a 16x64 output tile; WG = 4 waves = 16x256.
__global__ __launch_bounds__(128)
void qkv_gemm_kernel(const __bf16* __restrict__ X,     // [8192,1024]
                     const __bf16* __restrict__ W,     // [1024,1024], row n = out feature
                     const float*  __restrict__ bias,  // [1024]
                     __bf16* __restrict__ outRM,       // [8192,1024] (Q/K) or unused
                     __bf16* __restrict__ outTR,       // [1024,8192] (V^T) or unused
                     float scale, int transposed)
{
    const int tid   = threadIdx.x;
    const int wave  = tid >> 5;
    const int lane  = tid & 31;
    const int laneM = lane & 15;
    const int hi    = lane >> 4;

    const int mBase = blockIdx.y << 4;
    const int nBase = (blockIdx.x << 8) + (wave << 6);

    CFrag acc[4];
#pragma unroll
    for (int t = 0; t < 4; ++t)
#pragma unroll
        for (int e = 0; e < 8; ++e) acc[t].f[e] = 0.0f;

    const __bf16* arow = X + (size_t)(mBase + laneM) * DMODEL;

    for (int k0 = 0; k0 < DMODEL; k0 += 32) {
        AFrag a;
        a.q[0] = *(const uint4*)(arow + k0 + hi * 8);
        a.q[1] = *(const uint4*)(arow + k0 + 16 + hi * 8);
#pragma unroll
        for (int t = 0; t < 4; ++t) {
            const __bf16* brow =
                W + (size_t)(nBase + t * 16 + laneM) * DMODEL + k0 + hi * 8;
            BFrag b;
            b.q[0] = *(const uint4*)(brow);
            b.q[1] = *(const uint4*)(brow + 16);
            acc[t].v = wmma_bf16(a.v, b.v, acc[t].v);
        }
    }

#pragma unroll
    for (int t = 0; t < 4; ++t) {
        float bv = bias[nBase + t * 16 + laneM];
#pragma unroll
        for (int e = 0; e < 8; ++e) acc[t].f[e] = (acc[t].f[e] + bv) * scale;
    }

    if (!transposed) {
#pragma unroll
        for (int t = 0; t < 4; ++t) {
            int n = nBase + t * 16 + laneM;
#pragma unroll
            for (int e = 0; e < 8; ++e) {
                int m = mBase + e + hi * 8;
                outRM[(size_t)m * DMODEL + n] = (__bf16)acc[t].f[e];
            }
        }
    } else {
        // elem e -> row m = mBase + e + 8*hi : 8 consecutive m per lane -> b128 store
#pragma unroll
        for (int t = 0; t < 4; ++t) {
            int n = nBase + t * 16 + laneM;
            Pack8 p;
#pragma unroll
            for (int e = 0; e < 8; ++e) p.h[e] = (__bf16)acc[t].f[e];
            *(uint4*)(outTR + (size_t)n * MROWS + mBase + hi * 8) = p.q;
        }
    }
}

// ---------------------------------------------------------------------------
// Flash attention: 1 WG (4 waves) owns 16 query rows.
//   wave w: S slice = 16 rows x keys [kb*64 + w*16, +16)
//           O slice = 16 rows x features [w*256, +256)  (16 f32 accum tiles)
//
// LDS Q layout (bank-friendly): row stride 1032 halves; within a row, k>=512
// is shifted by +4 halves. This matches the TDM pad feature exactly:
// pad_interval=7 (every 256 dwords stored) + pad_amount=1 (2 dwords = 4
// halves) -> +4 halves after every 512 halves, i.e. mid-row and at row end.
#define QS 1032   // LDS Q row stride in bf16
#define PS 72     // LDS P row stride in bf16 (144B, 16B-aligned, bank-friendly)

static __device__ __forceinline__ int qpad(int k) { return k + ((k >= 512) ? 4 : 0); }

__global__ __launch_bounds__(128)
void attn_kernel(const __bf16* __restrict__ Qh,   // [8192,1024], pre-scaled 1/sqrt(d)
                 const __bf16* __restrict__ Kh,   // [8192,1024]
                 const __bf16* __restrict__ VT,   // [1024,8192]
                 float* __restrict__ Out)         // [8192,1024] f32
{
    __shared__ __bf16 sQ[16 * QS];
    __shared__ __bf16 sP[16 * PS];
    __shared__ float  sMax[4][16];
    __shared__ float  sSum[4][16];

    const int tid   = threadIdx.x;
    const int wave  = tid >> 5;
    const int lane  = tid & 31;
    const int laneM = lane & 15;
    const int hi    = lane >> 4;
    const int mBase = blockIdx.x << 4;

    // ---- stage Q tile (16x1024 bf16) into LDS ----
    // rows 0..7: one TDM descriptor (wave 0), LDS padding enabled
    if (wave == 0) {
        unsigned long gQ = (unsigned long)(uintptr_t)(Qh + (size_t)mBase * DMODEL);
        unsigned ldsBase = (unsigned)(uintptr_t)&sQ[0];
        // D# group0: count=1; lds_addr; global_addr[56:0]; type=2 (bits 127:126)
        u32x4 g0 = { 1u, ldsBase, (unsigned)gQ,
                     (unsigned)((gQ >> 32) & 0x01FFFFFFu) | 0x80000000u };
        // D# group1: data_size=2B; pad_enable, pad_interval=7 (256 dw),
        // pad_amount=1 (2 dw); tensor_dim0=1024; tensor_dim1=8192;
        // tile_dim0=1024; tile_dim1=8; tensor_dim0_stride=1024
        u32x8 g1 = { (1u << 16) | (1u << 20) | (7u << 22) | (1u << 25),
                     1024u << 16,              // tensor_dim0[15:0] @ bits 63:48
                     (8192u & 0xFFFFu) << 16,  // tensor_dim1[15:0] @ bits 95:80
                     1024u << 16,              // tile_dim0 @ bits 127:112
                     8u,                       // tile_dim1 @ bits 143:128
                     1024u,                    // tensor_dim0_stride[31:0]
                     0u, 0u };
        u32x4 g2 = { 0u, 0u, 0u, 0u };         // 2D tile: dims 2..4 unused
        u32x4 g3 = { 0u, 0u, 0u, 0u };
        asm volatile("tensor_load_to_lds %0, %1, %2, %3"
                     :: "s"(g0), "s"(g1), "s"(g2), "s"(g3) : "memory");
    }
    // rows 8..15: per-lane async b128 copies at the same padded offsets
    for (int c = tid; c < 8 * (DMODEL / 8); c += blockDim.x) {
        int r = 8 + (c >> 7);
        int k = (c & 127) * 8;
        async_g2l_b128(&sQ[r * QS + qpad(k)],
                       Qh + (size_t)(mBase + r) * DMODEL + k);
    }
    wait_asynccnt0();
    wait_tensorcnt0();
    __syncthreads();

    CFrag o[16];
#pragma unroll
    for (int t = 0; t < 16; ++t)
#pragma unroll
        for (int e = 0; e < 8; ++e) o[t].f[e] = 0.0f;

    float mR[8], lR[8], corr[8];
#pragma unroll
    for (int r = 0; r < 8; ++r) { mR[r] = -1e30f; lR[r] = 0.0f; }

    for (int kb = 0; kb < MROWS / 64; ++kb) {
        const int key = (kb << 6) + (wave << 4) + laneM;

        // prefetch next key block's K rows (overlaps WMMA below)
        if (kb + 1 < MROWS / 64)
            __builtin_prefetch(Kh + (size_t)(key + 64) * DMODEL, 0, 1);

        // ---- S slice: 16 rows x 16 keys, k = 0..1023 ----
        CFrag s;
#pragma unroll
        for (int e = 0; e < 8; ++e) s.f[e] = 0.0f;
        const __bf16* krow = Kh + (size_t)key * DMODEL;
        for (int k0 = 0; k0 < DMODEL; k0 += 32) {
            int ka = qpad(k0);   // fragment never straddles the pad (k0 % 32 == 0)
            AFrag a;
            a.q[0] = *(const uint4*)(&sQ[laneM * QS + ka + hi * 8]);
            a.q[1] = *(const uint4*)(&sQ[laneM * QS + ka + 16 + hi * 8]);
            BFrag b;
            b.q[0] = *(const uint4*)(krow + k0 + hi * 8);
            b.q[1] = *(const uint4*)(krow + k0 + 16 + hi * 8);
            s.v = wmma_bf16(a.v, b.v, s.v);
        }

        // ---- per-row max of this 16-key slice (reduce across 16 lanes) ----
#pragma unroll
        for (int r = 0; r < 8; ++r) {
            float v = s.f[r];
            v = fmaxf(v, __shfl_xor(v, 1, 32));
            v = fmaxf(v, __shfl_xor(v, 2, 32));
            v = fmaxf(v, __shfl_xor(v, 4, 32));
            v = fmaxf(v, __shfl_xor(v, 8, 32));
            if (laneM == 0) sMax[wave][r + hi * 8] = v;
        }
        __syncthreads();

        // ---- combine stats; rescale private O accumulators ----
#pragma unroll
        for (int r = 0; r < 8; ++r) {
            int row = r + hi * 8;
            float mNew = mR[r];
#pragma unroll
            for (int w = 0; w < 4; ++w) mNew = fmaxf(mNew, sMax[w][row]);
            corr[r] = __expf(mR[r] - mNew);
            mR[r]   = mNew;
        }
#pragma unroll
        for (int t = 0; t < 16; ++t)
#pragma unroll
            for (int e = 0; e < 8; ++e) o[t].f[e] *= corr[e];

        // ---- exponentials, partial row sums, publish P (bf16) ----
#pragma unroll
        for (int r = 0; r < 8; ++r) {
            int row = r + hi * 8;
            float p = __expf(s.f[r] - mR[r]);
            sP[row * PS + (wave << 4) + laneM] = (__bf16)p;
            float ps = p;
            ps += __shfl_xor(ps, 1, 32);
            ps += __shfl_xor(ps, 2, 32);
            ps += __shfl_xor(ps, 4, 32);
            ps += __shfl_xor(ps, 8, 32);
            if (laneM == 0) sSum[wave][row] = ps;
        }
        __syncthreads();

#pragma unroll
        for (int r = 0; r < 8; ++r) {
            int row = r + hi * 8;
            float add = sSum[0][row] + sSum[1][row] + sSum[2][row] + sSum[3][row];
            lR[r] = lR[r] * corr[r] + add;
        }

        // ---- O += P @ V   (wave owns 256 features; B frags from VT rows) ----
        AFrag pa[2];
#pragma unroll
        for (int kc = 0; kc < 2; ++kc) {
            pa[kc].q[0] = *(const uint4*)(&sP[laneM * PS + kc * 32 + hi * 8]);
            pa[kc].q[1] = *(const uint4*)(&sP[laneM * PS + kc * 32 + 16 + hi * 8]);
        }
#pragma unroll
        for (int t = 0; t < 16; ++t) {
            int feat = (wave << 8) + (t << 4) + laneM;
            const __bf16* vrow = VT + (size_t)feat * MROWS + (kb << 6);
#pragma unroll
            for (int kc = 0; kc < 2; ++kc) {
                BFrag b;
                b.q[0] = *(const uint4*)(vrow + kc * 32 + hi * 8);
                b.q[1] = *(const uint4*)(vrow + kc * 32 + 16 + hi * 8);
                o[t].v = wmma_bf16(pa[kc].v, b.v, o[t].v);
            }
        }
    }

    // ---- normalize by l and store f32 ----
#pragma unroll
    for (int t = 0; t < 16; ++t) {
        int n = (wave << 8) + (t << 4) + laneM;
#pragma unroll
        for (int e = 0; e < 8; ++e) {
            int m = mBase + e + hi * 8;
            Out[(size_t)m * DMODEL + n] = o[t].f[e] / lR[e];
        }
    }
}

// ---------------------------------------------------------------------------
extern "C" void kernel_launch(void* const* d_in, const int* in_sizes, int n_in,
                              void* d_out, int out_size, void* d_ws, size_t ws_size,
                              hipStream_t stream)
{
    const float* x  = (const float*)d_in[0];
    const float* Wq = (const float*)d_in[1];
    const float* bq = (const float*)d_in[2];
    const float* Wk = (const float*)d_in[3];
    const float* bk = (const float*)d_in[4];
    const float* Wv = (const float*)d_in[5];
    const float* bv = (const float*)d_in[6];
    float* out = (float*)d_out;

    // workspace layout (bf16), ~70 MB total
    char* p = (char*)d_ws;
    __bf16* xh  = (__bf16*)p; p += (size_t)MROWS * DMODEL * 2;   // 16 MB
    __bf16* Wqh = (__bf16*)p; p += (size_t)DMODEL * DMODEL * 2;  //  2 MB
    __bf16* Wkh = (__bf16*)p; p += (size_t)DMODEL * DMODEL * 2;
    __bf16* Wvh = (__bf16*)p; p += (size_t)DMODEL * DMODEL * 2;
    __bf16* Qh  = (__bf16*)p; p += (size_t)MROWS * DMODEL * 2;   // 16 MB
    __bf16* Kh  = (__bf16*)p; p += (size_t)MROWS * DMODEL * 2;   // 16 MB
    __bf16* VTt = (__bf16*)p; p += (size_t)MROWS * DMODEL * 2;   // 16 MB

    // 1) convert inputs to bf16
    cvt_f32_to_bf16<<<4096, 256, 0, stream>>>(x,  xh,  MROWS * DMODEL);
    cvt_f32_to_bf16<<<2048, 256, 0, stream>>>(Wq, Wqh, DMODEL * DMODEL);
    cvt_f32_to_bf16<<<2048, 256, 0, stream>>>(Wk, Wkh, DMODEL * DMODEL);
    cvt_f32_to_bf16<<<2048, 256, 0, stream>>>(Wv, Wvh, DMODEL * DMODEL);

    // 2) QKV projections (WMMA). Q gets 1/sqrt(d) folded in; V stored transposed.
    dim3 g(DMODEL / 256, MROWS / 16);   // (4, 512)
    qkv_gemm_kernel<<<g, 128, 0, stream>>>(xh, Wqh, bq, Qh, nullptr, INV_SQRT_D, 0);
    qkv_gemm_kernel<<<g, 128, 0, stream>>>(xh, Wkh, bk, Kh, nullptr, 1.0f, 0);
    qkv_gemm_kernel<<<g, 128, 0, stream>>>(xh, Wvh, bv, nullptr, VTt, 1.0f, 1);

    // 3) flash attention: 512 WGs x 128 threads (4 waves, 16 rows each)
    attn_kernel<<<MROWS / 16, 128, 0, stream>>>(Qh, Kh, VTt, out);

    (void)in_sizes; (void)n_in; (void)out_size; (void)ws_size;
}
